// LSTMFromEmbeddings_4097398800771
// MI455X (gfx1250) — matrix operand order repaired and checked by reading, not compiled, not run on hardware
//
#include <hip/hip_runtime.h>
#include <hip/hip_bf16.h>

// Problem constants (from the reference): B=128, T=1024, E=256, H=256, C=2
#define Bsz    128
#define Tlen   1024
#define Edim   256
#define Hdim   256
#define FOURH  1024   // 4*H
#define KTOT   512    // E + H (fused GEMM K)
#define Cdim   2

typedef __attribute__((ext_vector_type(16))) _Float16 v16h;
typedef __attribute__((ext_vector_type(8)))  _Float16 v8h;
typedef __attribute__((ext_vector_type(8)))  float    v8f;

// fast device transcendentals (hardware v_exp + fast divide)
__device__ __forceinline__ float sigf(float x) {
    return __fdividef(1.0f, 1.0f + __expf(-x));
}
__device__ __forceinline__ float tanh_fast(float x) {
    float e = __expf(2.0f * x);
    return __fdividef(e - 1.0f, e + 1.0f);
}

// ---------------------------------------------------------------------------
// Kernel 1: xh = f16(embeddings * mask), layout [B, T, E]
// ---------------------------------------------------------------------------
__global__ void k_convert_x(const float* __restrict__ emb,
                            const float* __restrict__ mask,
                            _Float16* __restrict__ xh) {
    long i8 = (long)blockIdx.x * blockDim.x + threadIdx.x;   // one 8-elem chunk
    long base = i8 * 8;
    if (base >= (long)Bsz * Tlen * Edim) return;
    // E % 8 == 0, so the whole chunk shares one (b,t) -> one mask value
    long bt = base / Edim;              // flat b*T + t
    float mv = mask[bt];
    const float4* s = (const float4*)(emb + base);
    float4 a0 = s[0], a1 = s[1];
    v8h o;
    o[0] = (_Float16)(a0.x * mv); o[1] = (_Float16)(a0.y * mv);
    o[2] = (_Float16)(a0.z * mv); o[3] = (_Float16)(a0.w * mv);
    o[4] = (_Float16)(a1.x * mv); o[5] = (_Float16)(a1.y * mv);
    o[6] = (_Float16)(a1.z * mv); o[7] = (_Float16)(a1.w * mv);
    *(v8h*)(xh + base) = o;
}

// ---------------------------------------------------------------------------
// Kernel 2: fused weights Wc[dir][n][k], k<256 -> W_ih[n][k], else W_hh[n][k-256]
// (f16, row-contiguous in K so each lane's B-fragment is two 16B loads)
// ---------------------------------------------------------------------------
__global__ void k_convert_w(const float* __restrict__ Wih_f, const float* __restrict__ Whh_f,
                            const float* __restrict__ Wih_b, const float* __restrict__ Whh_b,
                            _Float16* __restrict__ Wc) {
    int i = blockIdx.x * blockDim.x + threadIdx.x;  // 0 .. 2*1024*512-1
    if (i >= 2 * FOURH * KTOT) return;
    int dir = i >> 19;                 // 1024*512 = 2^19
    int r   = i & ((1 << 19) - 1);
    int n   = r >> 9;                  // / KTOT
    int k   = r & (KTOT - 1);
    const float* Wih = dir ? Wih_b : Wih_f;
    const float* Whh = dir ? Whh_b : Whh_f;
    float v = (k < Edim) ? Wih[n * Edim + k] : Whh[n * Hdim + (k - Edim)];
    Wc[i] = (_Float16)v;
}

// ---------------------------------------------------------------------------
// Kernel 3: the recurrence. 16 workgroups: blockIdx/8 = direction, %8 = batch
// row-block of 16. 512 threads = 16 waves; wave w owns gate columns
// [16w, 16w+16) of all four gate blocks (i,f,g,o) -> cell update in registers.
// h and x tiles are double-buffered in LDS -> ONE s_barrier per timestep.
// Per step, per wave: 16 K-iterations x 4 tiles = 64 chained WMMAs.
// ---------------------------------------------------------------------------
__global__ __launch_bounds__(512)
void k_lstm(const _Float16* __restrict__ xh,
            const _Float16* __restrict__ Wc,
            const float* __restrict__ b_f, const float* __restrict__ b_b,
            float* __restrict__ hstates) {
    // +8 f16 pad per row: row stride 528B = 132 dwords -> 4-bank skew per row,
    // so the 16 rows of an A-fragment read hit distinct banks.
    __shared__ _Float16 xL[2][16][Edim + 8];
    __shared__ _Float16 hL[2][16][Hdim + 8];

    const int dir    = blockIdx.x >> 3;
    const int rowblk = blockIdx.x & 7;
    const int m0     = rowblk * 16;
    const _Float16* Wd  = Wc + (size_t)dir * FOURH * KTOT;
    const float*    bia = dir ? b_b : b_f;

    const int tid  = threadIdx.x;
    const int wave = tid >> 5;
    const int lane = tid & 31;
    const int lcol = lane & 15;     // column within 16x16 tile (B/C/D layout)
    const int lhi  = lane >> 4;     // lane half
    const int n16  = wave * 16;     // this wave's gate-column base
    const int nc   = n16 + lcol;    // this lane's gate column (0..255 within block)
    const int mrb  = lhi * 8;       // D rows: mrb + r, r = 0..7
    // A-fragment addressing (16x32 f16 tile, ISA layout):
    const int arow = lcol;          // lanes 0-15 rows 0-15 (lanes 16-31 same rows)
    const int akb  = lhi * 8;       // K sub-block: lo half K 0..7/16..23, hi half 8..15/24..31
    // per-lane weight-row base; gate g adds g*256*KTOT (folds into imm offset)
    const size_t lane_w = (size_t)nc * KTOT;

    // zero h ping buffer (read by step 0)
    for (int i = tid; i < 16 * (Hdim + 8); i += 512) ((_Float16*)hL[0])[i] = (_Float16)0.0f;

    // x-tile loader: 512 threads move 16 rows x 256 f16 (one v8h each)
    const int xrow = tid >> 5;
    const int xoff = (tid & 31) * 8;
    const size_t xrow_base = ((size_t)(m0 + xrow) * Tlen) * Edim + xoff;

    { // first timestep's x tile into ping buffer
        const int t0 = dir ? (Tlen - 1) : 0;
        *(v8h*)&xL[0][xrow][xoff] = *(const v8h*)(xh + xrow_base + (size_t)t0 * Edim);
    }

    // biases are step-invariant: hoist
    const float bI = bia[nc], bF = bia[Hdim + nc], bG = bia[2 * Hdim + nc], bO = bia[3 * Hdim + nc];

    v8f creg = {0.f, 0.f, 0.f, 0.f, 0.f, 0.f, 0.f, 0.f};
    v8f hsum = {0.f, 0.f, 0.f, 0.f, 0.f, 0.f, 0.f, 0.f};

    __syncthreads();

    for (int s = 0; s < Tlen; ++s) {
        const int cur = s & 1, nxt = cur ^ 1;

        // Opaque per-step OFFSET (not pointer!): LLVM cannot hoist the 64
        // B-fragment loads out of the time loop (no 512-VGPR live range, no
        // scratch spills), while the base pointer keeps its inferred global
        // address space -> global_load_b128 from L2, decoupled from DScnt.
        unsigned wofs = 0;
        asm volatile("" : "+s"(wofs));
        const _Float16* Wl = Wd + lane_w + wofs;

        v8f acc0 = {0.f,0.f,0.f,0.f,0.f,0.f,0.f,0.f};  // i-gate tile
        v8f acc1 = {0.f,0.f,0.f,0.f,0.f,0.f,0.f,0.f};  // f-gate tile
        v8f acc2 = {0.f,0.f,0.f,0.f,0.f,0.f,0.f,0.f};  // g-gate tile
        v8f acc3 = {0.f,0.f,0.f,0.f,0.f,0.f,0.f,0.f};  // o-gate tile

        #pragma unroll
        for (int k0 = 0; k0 < KTOT; k0 += 32) {
            // A fragment: shared across the 4 gate tiles (operand reuse)
            const _Float16* Asrc = (k0 < Edim) ? &xL[cur][arow][k0 + akb]
                                               : &hL[cur][arow][(k0 - Edim) + akb];
            v8h alo = *(const v8h*)(Asrc);        // K = k0+akb .. +7
            v8h ahi = *(const v8h*)(Asrc + 16);   // K = k0+16+akb .. +7
            v16h A  = __builtin_shufflevector(alo, ahi,
                        0,1,2,3,4,5,6,7,8,9,10,11,12,13,14,15);
            // B fragments: K = k0+akb..+7 and k0+16+akb..+7 of Wcat row (col of B = W row);
            // gate offsets are compile-time immediates (g*256KB + 2*k0 bytes).
            v16h B0 = __builtin_shufflevector(*(const v8h*)(Wl + 0 * Hdim * KTOT + k0 + akb), *(const v8h*)(Wl + 0 * Hdim * KTOT + k0 + 16 + akb), 0,1,2,3,4,5,6,7,8,9,10,11,12,13,14,15);
            v16h B1 = __builtin_shufflevector(*(const v8h*)(Wl + 1 * Hdim * KTOT + k0 + akb), *(const v8h*)(Wl + 1 * Hdim * KTOT + k0 + 16 + akb), 0,1,2,3,4,5,6,7,8,9,10,11,12,13,14,15);
            v16h B2 = __builtin_shufflevector(*(const v8h*)(Wl + 2 * Hdim * KTOT + k0 + akb), *(const v8h*)(Wl + 2 * Hdim * KTOT + k0 + 16 + akb), 0,1,2,3,4,5,6,7,8,9,10,11,12,13,14,15);
            v16h B3 = __builtin_shufflevector(*(const v8h*)(Wl + 3 * Hdim * KTOT + k0 + akb), *(const v8h*)(Wl + 3 * Hdim * KTOT + k0 + 16 + akb), 0,1,2,3,4,5,6,7,8,9,10,11,12,13,14,15);
            acc0 = __builtin_amdgcn_wmma_f32_16x16x32_f16(false, A, false, B0, (short)0, acc0, false, false);
            acc1 = __builtin_amdgcn_wmma_f32_16x16x32_f16(false, A, false, B1, (short)0, acc1, false, false);
            acc2 = __builtin_amdgcn_wmma_f32_16x16x32_f16(false, A, false, B2, (short)0, acc2, false, false);
            acc3 = __builtin_amdgcn_wmma_f32_16x16x32_f16(false, A, false, B3, (short)0, acc3, false, false);
        }

        // LSTM cell update, fully in registers (D layout: lane holds col nc,
        // rows mrb..mrb+7). Writes go to the *other* h buffer, so no barrier
        // is needed between the GEMM reads and these writes.
        #pragma unroll
        for (int r = 0; r < 8; ++r) {
            float gi = sigf(acc0[r] + bI);
            float gf = sigf(acc1[r] + bF);
            float gg = tanh_fast(acc2[r] + bG);
            float go = sigf(acc3[r] + bO);
            float cv = gf * creg[r] + gi * gg;
            creg[r] = cv;
            float hv = go * tanh_fast(cv);
            hsum[r] += hv;
            hL[nxt][mrb + r][nc] = (_Float16)hv;   // publish h for next step
        }

        // stage next x tile into the other buffer (+ prefetch one step ahead)
        if (s + 1 < Tlen) {
            const int tn = dir ? (Tlen - 2 - s) : (s + 1);
            *(v8h*)&xL[nxt][xrow][xoff] = *(const v8h*)(xh + xrow_base + (size_t)tn * Edim);
            if (s + 2 < Tlen) {
                const int tp = dir ? (Tlen - 3 - s) : (s + 2);
                __builtin_prefetch((const void*)(xh + xrow_base + (size_t)tp * Edim), 0, 1);
            }
        }
        __syncthreads();   // single barrier per step (ping-pong buffers)
    }

    // mean pool -> hstates[m][dir*H + n]  (f32, [B, 2H])
    const float inv_t = 1.0f / (float)Tlen;
    #pragma unroll
    for (int r = 0; r < 8; ++r)
        hstates[(size_t)(m0 + mrb + r) * (2 * Hdim) + dir * Hdim + nc] = hsum[r] * inv_t;
}

// ---------------------------------------------------------------------------
// Kernel 4: out = hstates @ W_linT + b_lin   ([128,512] x [512,2] -> tiny)
// ---------------------------------------------------------------------------
__global__ void k_head(const float* __restrict__ hstates,
                       const float* __restrict__ Wlin,
                       const float* __restrict__ blin,
                       float* __restrict__ out) {
    int tid = blockIdx.x * blockDim.x + threadIdx.x;
    if (tid >= Bsz * Cdim) return;
    int m = tid >> 1, c = tid & 1;
    float acc = blin[c];
    const float* hv = hstates + (size_t)m * (2 * Hdim);
    const float* wv = Wlin + (size_t)c * (2 * Hdim);
    #pragma unroll 8
    for (int k = 0; k < 2 * Hdim; ++k) acc = fmaf(hv[k], wv[k], acc);
    out[tid] = acc;
}

// ---------------------------------------------------------------------------
extern "C" void kernel_launch(void* const* d_in, const int* in_sizes, int n_in,
                              void* d_out, int out_size, void* d_ws, size_t ws_size,
                              hipStream_t stream) {
    const float* emb   = (const float*)d_in[0];
    const float* mask  = (const float*)d_in[1];
    const float* Wih_f = (const float*)d_in[2];
    const float* Whh_f = (const float*)d_in[3];
    const float* b_f   = (const float*)d_in[4];
    const float* Wih_b = (const float*)d_in[5];
    const float* Whh_b = (const float*)d_in[6];
    const float* b_b   = (const float*)d_in[7];
    const float* Wlin  = (const float*)d_in[8];
    const float* blin  = (const float*)d_in[9];
    float* out = (float*)d_out;

    // workspace carve (256B aligned)
    char* ws = (char*)d_ws;
    size_t off = 0;
    auto carve = [&](size_t bytes) -> void* {
        void* p = ws + off;
        off += (bytes + 255) & ~(size_t)255;
        return p;
    };
    _Float16* xh      = (_Float16*)carve((size_t)Bsz * Tlen * Edim * sizeof(_Float16)); // 64 MB
    _Float16* Wc      = (_Float16*)carve((size_t)2 * FOURH * KTOT * sizeof(_Float16));  // 2 MB
    float*    hstates = (float*)   carve((size_t)Bsz * 2 * Hdim * sizeof(float));       // 256 KB
    (void)ws_size; (void)in_sizes; (void)n_in; (void)out_size;

    {   // 1) mask + convert embeddings -> f16
        long n8 = (long)Bsz * Tlen * Edim / 8;
        k_convert_x<<<(unsigned)((n8 + 255) / 256), 256, 0, stream>>>(emb, mask, xh);
    }
    {   // 2) fuse + convert weights -> f16
        int n = 2 * FOURH * KTOT;
        k_convert_w<<<(n + 255) / 256, 256, 0, stream>>>(Wih_f, Whh_f, Wih_b, Whh_b, Wc);
    }
    // 3) bidirectional recurrence: 8 row-block WGs per direction, no grid sync
    k_lstm<<<16, 512, 0, stream>>>(xh, Wc, b_f, b_b, hstates);
    // 4) linear head
    k_head<<<1, 256, 0, stream>>>(hstates, Wlin, blin, out);
}
